// SRLSecondOrderScorer_44659069943940
// MI455X (gfx1250) — matrix operand order repaired and checked by reading, not compiled, not run on hardware
//
#include <hip/hip_runtime.h>

typedef __bf16 bf16;
typedef __attribute__((ext_vector_type(16))) __bf16 v16bf;
typedef __attribute__((ext_vector_type(8)))  __bf16 v8bf;
typedef __attribute__((ext_vector_type(8)))  float  v8f;
typedef __attribute__((ext_vector_type(4)))  unsigned v4u;
typedef __attribute__((ext_vector_type(8)))  int      v8i;
typedef __attribute__((ext_vector_type(4)))  int      v4i;

// ---------- helpers ----------

__device__ __forceinline__ bf16 f2bf(float f) {
    union { float f; unsigned u; } v; v.f = f;
    unsigned r = (v.u + 0x7FFFu + ((v.u >> 16) & 1u)) >> 16;
    unsigned short s = (unsigned short)r;
    return __builtin_bit_cast(bf16, s);
}

__device__ __forceinline__ v8f v8f_zero() {
    v8f z;
#pragma unroll
    for (int i = 0; i < 8; ++i) z[i] = 0.0f;
    return z;
}

// A-operand per-lane vector for v_wmma_f32_16x16x32_bf16.
// lane = m + 16*hi holds row m: elems 0..15 = row[kc+16*hi .. +16)
__device__ __forceinline__ v16bf ld_a(const bf16* row, int kc, int hi) {
    v8bf a = *(const v8bf*)(row + kc + 16 * hi);
    v8bf b = *(const v8bf*)(row + kc + 16 * hi + 8);
    return __builtin_shufflevector(a, b, 0,1,2,3,4,5,6,7,8,9,10,11,12,13,14,15);
}

// B-operand: 16 contiguous bf16 (reduction elements kc+16*hi .. +16) for this
// lane's column; p must be 16B aligned.
__device__ __forceinline__ v16bf ld_b(const bf16* p) {
    v8bf a = *(const v8bf*)p;
    v8bf b = *(const v8bf*)(p + 8);
    return __builtin_shufflevector(a, b, 0,1,2,3,4,5,6,7,8,9,10,11,12,13,14,15);
}

__device__ __forceinline__ v8f wmma_bf16(v16bf a, v16bf b, v8f c) {
    return __builtin_amdgcn_wmma_f32_16x16x32_bf16(false, a, false, b,
                                                   (short)0, c, false, false);
}

// TDM 1-D global->LDS copy of `ndw` dwords. D# per CDNA5 ISA:
// group0: count=1 | lds_addr | global_addr[56:0] | type=2
// group1: data_size=4B, tensor_dim0=tile_dim0=ndw, tensor_dim1=1, strides=ndw
// Issue from a single wave; complete with s_wait_tensorcnt.
// This toolchain exposes the 6-arg builtin (extra int32x8 group before cpol).
__device__ __forceinline__ void tdm_copy_1d(const void* gsrc, void* ldsdst,
                                            unsigned ndw) {
    unsigned long long ga = (unsigned long long)gsrc;
    unsigned lds = (unsigned)(unsigned long long)ldsdst;  // LDS offset = addr[31:0]
    v4u g0;
    g0[0] = 1u;                                            // count=1 (user D#)
    g0[1] = lds;                                           // lds_addr
    g0[2] = (unsigned)ga;                                  // global_addr[31:0]
    g0[3] = (unsigned)((ga >> 32) & 0x01FFFFFFull) | (2u << 30);  // ga[56:32], type=2
    unsigned td0 = ndw, td1 = 1u, tile0 = ndw;
    unsigned long long s0 = ndw, s1 = ndw;
    v8i g1;
    g1[0] = (int)(2u << 16);                               // data_size = 4 bytes
    g1[1] = (int)((td0 & 0xFFFFu) << 16);                  // tensor_dim0[15:0]
    g1[2] = (int)((td0 >> 16) | ((td1 & 0xFFFFu) << 16));  // td0[31:16], td1[15:0]
    g1[3] = (int)((td1 >> 16) | (tile0 << 16));            // td1[31:16], tile_dim0
    g1[4] = 0;                                             // tile_dim1=0, tile_dim2=0
    g1[5] = (int)(unsigned)(s0 & 0xFFFFFFFFull);           // stride0[31:0]
    g1[6] = (int)((unsigned)(s0 >> 32) | (((unsigned)s1 & 0xFFFFu) << 16));
    g1[7] = (int)((unsigned)(s1 >> 16));                   // stride1[47:16]
    v4i g2; g2[0] = g2[1] = g2[2] = g2[3] = 0;
    v4i g3; g3[0] = g3[1] = g3[2] = g3[3] = 0;
    v8i g4;
#pragma unroll
    for (int i = 0; i < 8; ++i) g4[i] = 0;
    __builtin_amdgcn_tensor_load_to_lds(g0, g1, g2, g3, g4, 0);
}

// ---------- constants ----------
// D=150 padded to KP=160 (5 chunks of K=32). N=128, B=2, rows BX=256.
#define KP   160
#define A1_ROWSZ (150 * KP)              // per (b,x) A1 slab, elements
#define OUT_T (2u * 128u * 128u * 128u)  // 4194304 elements per output tensor

// ---------- kernel 1: MLP + LeakyReLU + bf16 pack (zero-padded to KP) ----------

__global__ __launch_bounds__(256) void mlp_kernel(
    const float* __restrict__ x,
    const float* __restrict__ W_sh, const float* __restrict__ b_sh,
    const float* __restrict__ W_st, const float* __restrict__ b_st,
    const float* __restrict__ W_p,  const float* __restrict__ b_p,
    bf16* __restrict__ H)            // 3 matrices of [256][KP]
{
    int row = blockIdx.x;            // 0..255 (= b*128 + n)
    int m   = blockIdx.y;            // 0..2 -> sh, st, p
    const float* W    = (m == 0) ? W_sh : ((m == 1) ? W_st : W_p);
    const float* bias = (m == 0) ? b_sh : ((m == 1) ? b_st : b_p);
    bf16* Hm = H + m * (256 * KP);

    __shared__ float xr[1024];
    for (int i = threadIdx.x; i < 1024; i += 256) xr[i] = x[row * 1024 + i];
    __syncthreads();

    int k = threadIdx.x;
    if (k < KP) {
        float acc = 0.0f;
        if (k < 150) {
            acc = bias[k];
#pragma unroll 4
            for (int i = 0; i < 1024; ++i) acc += xr[i] * W[i * 150 + k];
            acc = (acc > 0.0f) ? acc : 0.1f * acc;   // LeakyReLU(0.1)
        }
        Hm[row * KP + k] = f2bf(acc);
    }
}

// ---------- kernel 2 (step A): A1t[b,x][j][k] = sum_i X[b,x,i] * T[i,k,j] ----------
// grid: (j = 0..149, half = 0..1); block 256 = 8 waves, 16 rows(bx)/wave.

__global__ __launch_bounds__(256) void stepA_kernel(
    const float* __restrict__ T,     // [150][150][150], T[i][k][j]
    const bf16*  __restrict__ Hx,    // [256][KP]
    bf16* __restrict__ A1t)          // [256][150][KP]
{
    int j    = blockIdx.x;
    int half = blockIdx.y;
    int tid  = threadIdx.x;
    int lane = tid & 31, wav = tid >> 5;
    int lo = lane & 15, hi = lane >> 4;

    __shared__ bf16 ldsB[KP * 40];   // [k][i_local], row stride 40 (80B, 16B mult)

    const bf16* arow = Hx + (half * 128 + wav * 16 + lo) * KP;

    v8f acc[10];
#pragma unroll
    for (int c = 0; c < 10; ++c) acc[c] = v8f_zero();

    for (int ic = 0; ic < 5; ++ic) {                 // i chunks of 32
        __syncthreads();
        int ibase = ic * 32;
        for (int u = tid; u < KP * 32; u += 256) {   // stage T chunk, transposed, ->bf16
            int k  = u % KP;
            int il = u / KP;
            int ig = ibase + il;
            float v = 0.0f;
            if (k < 150 && ig < 150) v = T[ig * 22500 + k * 150 + j];
            ldsB[k * 40 + il] = f2bf(v);
        }
        __syncthreads();

        // speculative prefetch of the next i-chunk of T into L2 (hides the
        // strided-gather latency behind this chunk's WMMAs)
        if (ic + 1 < 5) {
            int ip = (ic + 1) * 32 + (tid >> 3);
            int kp = (tid & 7) * 20;
            if (ip < 150 && kp < 150)
                __builtin_prefetch(&T[ip * 22500 + kp * 150 + j], 0, 0);
        }

        v16bf a = ld_a(arow, ic * 32, hi);
#pragma unroll
        for (int c = 0; c < 10; ++c) {
            v16bf b = ld_b(&ldsB[(c * 16 + lo) * 40 + hi * 16]);
            acc[c] = wmma_bf16(a, b, acc[c]);
        }
    }

    int row0 = half * 128 + wav * 16;
#pragma unroll
    for (int c = 0; c < 10; ++c) {
        int k = c * 16 + lo;
#pragma unroll
        for (int r = 0; r < 8; ++r) {
            int row = row0 + r + 8 * hi;
            A1t[row * A1_ROWSZ + j * KP + k] = f2bf(acc[c][r]);
        }
    }
}

// ---------- kernel 3 (steps B+C fused): one WG per (b,x) ----------
// C[z,j] = Hz[b] @ A1[b,x]  (reduce k) ; S[z,y] = C @ Hy[b]^T (reduce j)
// dest[b][z][x][y]. Staging via Tensor Data Mover.

__global__ __launch_bounds__(256) void stepBC_kernel(
    const bf16* __restrict__ A1t,    // [256][150][KP]
    const bf16* __restrict__ Hz,     // [256][KP]
    const bf16* __restrict__ Hy,     // [256][KP]
    float* __restrict__ dest)
{
    int xi = blockIdx.x;             // 0..127
    int b  = blockIdx.y;             // 0..1
    int tid = threadIdx.x;
    int lane = tid & 31, wav = tid >> 5;
    int lo = lane & 15, hi = lane >> 4;

    extern __shared__ char smem[];
    bf16* ldsA1 = (bf16*)smem;                 // [160][KP]  (rows 150..159 zero)
    bf16* ldsC  = ldsA1 + 160 * KP;            // [128][KP]
    bf16* ldsY  = ldsC  + 128 * KP;            // [128][KP]

    // TDM stage: A1[b,x] slab (150*KP bf16 = 12000 dwords) and Hy[b]
    // (128*KP bf16 = 10240 dwords). One wave issues; TENSORcnt completes;
    // the workgroup barrier publishes LDS to all waves.
    if (tid < 32) {
        tdm_copy_1d(A1t + (size_t)(b * 128 + xi) * A1_ROWSZ, ldsA1, (150 * KP) / 2);
        tdm_copy_1d(Hy + b * 128 * KP, ldsY, (128 * KP) / 2);
        __builtin_amdgcn_s_wait_tensorcnt(0);
    }
    // zero pad rows 150..159 of ldsA1 (10 rows * 320B = 800 dwords)
    for (int u = tid; u < 800; u += 256)
        ((unsigned*)(ldsA1 + 150 * KP))[u] = 0u;
    __syncthreads();

    // ---- phase 1: C[z,j] ----
    const bf16* zrow = Hz + (b * 128 + wav * 16 + lo) * KP;
    v8f acc[10];
#pragma unroll
    for (int c = 0; c < 10; ++c) acc[c] = v8f_zero();
#pragma unroll
    for (int kc = 0; kc < 5; ++kc) {
        v16bf a = ld_a(zrow, kc * 32, hi);
#pragma unroll
        for (int c = 0; c < 10; ++c) {
            v16bf bm = ld_b(&ldsA1[(c * 16 + lo) * KP + kc * 32 + hi * 16]);
            acc[c] = wmma_bf16(a, bm, acc[c]);
        }
    }
#pragma unroll
    for (int c = 0; c < 10; ++c)
#pragma unroll
        for (int r = 0; r < 8; ++r)
            ldsC[(wav * 16 + r + 8 * hi) * KP + c * 16 + lo] = f2bf(acc[c][r]);
    __syncthreads();

    // ---- phase 2: S[z,y] ----
    v8f acc2[8];
#pragma unroll
    for (int c = 0; c < 8; ++c) acc2[c] = v8f_zero();
    const bf16* crow = &ldsC[(wav * 16 + lo) * KP];
#pragma unroll
    for (int jc = 0; jc < 5; ++jc) {
        v16bf a = ld_a(crow, jc * 32, hi);
#pragma unroll
        for (int c = 0; c < 8; ++c) {
            v16bf bm = ld_b(&ldsY[(c * 16 + lo) * KP + jc * 32 + hi * 16]);
            acc2[c] = wmma_bf16(a, bm, acc2[c]);
        }
    }

    float* out = dest + (size_t)b * 2097152 + (size_t)xi * 128;
#pragma unroll
    for (int c = 0; c < 8; ++c) {
#pragma unroll
        for (int r = 0; r < 8; ++r) {
            int z = wav * 16 + r + 8 * hi;
            out[(size_t)z * 16384 + c * 16 + lo] = acc2[c][r];  // lanes 0-15: 64B contig
        }
    }
}

// ---------- post passes ----------
// out[b,z,x,y] = s[b,z,min(x,y),max(x,y)]
__global__ __launch_bounds__(256) void sym_kernel(const float* __restrict__ s,
                                                  float* __restrict__ out) {
    unsigned id = blockIdx.x * 256u + threadIdx.x;
    int y = id & 127, xx = (id >> 7) & 127, zz = (id >> 14) & 127, b = id >> 21;
    int mn = min(xx, y), mx = max(xx, y);
    out[id] = s[(((unsigned)(b * 128 + zz) * 128u + mn) * 128u) + mx];
}

// out[b,x,y,z] = s[b,z,min(x,y),max(x,y)]   (triu_sym + permute(0,2,3,1))
__global__ __launch_bounds__(256) void symperm_kernel(const float* __restrict__ s,
                                                      float* __restrict__ out) {
    unsigned id = blockIdx.x * 256u + threadIdx.x;
    int zz = id & 127, y = (id >> 7) & 127, xx = (id >> 14) & 127, b = id >> 21;
    int mn = min(xx, y), mx = max(xx, y);
    out[id] = s[(((unsigned)(b * 128 + zz) * 128u + mn) * 128u) + mx];
}

// ---------- launcher ----------

extern "C" void kernel_launch(void* const* d_in, const int* in_sizes, int n_in,
                              void* d_out, int out_size, void* d_ws, size_t ws_size,
                              hipStream_t stream) {
    (void)in_sizes; (void)n_in; (void)out_size; (void)ws_size;

    const float* x    = (const float*)d_in[0];
    const float* W_sh = (const float*)d_in[1]; const float* b_sh = (const float*)d_in[2];
    const float* W_st = (const float*)d_in[3]; const float* b_st = (const float*)d_in[4];
    const float* W_p  = (const float*)d_in[5]; const float* b_p  = (const float*)d_in[6];

    // output slot order: span_psh, span_pst, ph_sib, pt_sib, ph_cop, pt_cop
    const float* T_slot[6] = {
        (const float*)d_in[8],   // T_ph    -> span_psh
        (const float*)d_in[7],   // T_pt    -> span_pst
        (const float*)d_in[9],   // T_phsib -> ph_sib
        (const float*)d_in[10],  // T_ptsib -> pt_sib
        (const float*)d_in[11],  // T_phcop -> ph_cop
        (const float*)d_in[12],  // T_ptcop -> pt_cop
    };
    // H indices: 0=sh, 1=st, 2=p ; triaffine(X, Y, Z, T)
    const int HX[6]   = {0, 0, 0, 1, 2, 2};
    const int HYi[6]  = {1, 1, 0, 1, 2, 2};
    const int HZ[6]   = {2, 2, 2, 2, 0, 1};
    const int POST[6] = {0, 1, 1, 1, 2, 2};  // 0=none, 1=sym, 2=sym+permute

    bf16*  H    = (bf16*)d_ws;                                  // 3 * 256*160 bf16
    bf16*  A1t  = (bf16*)((char*)d_ws + 262144);                // 256*150*160 bf16
    float* sraw = (float*)((char*)d_ws + 262144 + 12582912);    // 2*128^3 f32
    float* out  = (float*)d_out;

    mlp_kernel<<<dim3(256, 3), 256, 0, stream>>>(x, W_sh, b_sh, W_st, b_st, W_p, b_p, H);

    const size_t LDS_BC = (size_t)(160 * KP + 128 * KP + 128 * KP) * sizeof(bf16); // 133120

    for (int t = 0; t < 6; ++t) {
        stepA_kernel<<<dim3(150, 2), 256, 0, stream>>>(T_slot[t], H + HX[t] * (256 * KP), A1t);
        float* dst = (POST[t] == 0) ? (out + (size_t)t * OUT_T) : sraw;
        stepBC_kernel<<<dim3(128, 2), 256, LDS_BC, stream>>>(
            A1t, H + HZ[t] * (256 * KP), H + HYi[t] * (256 * KP), dst);
        if (POST[t] == 1)
            sym_kernel<<<32768, 256, 0, stream>>>(sraw, out + (size_t)t * OUT_T);
        else if (POST[t] == 2)
            symperm_kernel<<<32768, 256, 0, stream>>>(sraw, out + (size_t)t * OUT_T);
    }
}
